// SimplifiedEncoder_69621419868687
// MI455X (gfx1250) — matrix operand order repaired and checked by reading, not compiled, not run on hardware
//
#include <hip/hip_runtime.h>
#include <hip/hip_bf16.h>
#include <stdint.h>

#define SEQ   512
#define BATCH 128
#define DH    1024        // hidden size
#define KTOT  (2*DH)      // GEMM K = 2048 ([x|h])
#define KC    32          // K chunk per WMMA
#define NKC   (KTOT/KC)   // 64 chunks
#define SA    36          // LDS A row stride (bf16 elems, padded)
#define SB    36          // LDS B row stride: 32 data + 2 DW TDM pad (72 B)

typedef __attribute__((ext_vector_type(16))) __bf16 v16bf;
typedef __attribute__((ext_vector_type(8)))  float  v8f;
typedef __attribute__((ext_vector_type(4)))  unsigned int v4u;
typedef __attribute__((ext_vector_type(8)))  int    v8i;
typedef __attribute__((ext_vector_type(4)))  int    v4i;

union Frag { v16bf v; unsigned int u[8]; };

__device__ __forceinline__ unsigned short f2bf(float f) {
  unsigned int u = __float_as_uint(f);
  return (unsigned short)((u + 0x7FFFu + ((u >> 16) & 1u)) >> 16);   // RNE
}
__device__ __forceinline__ float sigm(float x) { return 1.f / (1.f + __expf(-x)); }
__device__ __forceinline__ float tanhfast(float x) {
  float e = __expf(-2.f * x); return (1.f - e) / (1.f + e);
}

// TDM 2-D tile load: 64 rows x 32 bf16, row stride 2048 elems in memory,
// LDS padding 2 DWORDs per 16-DWORD row -> 72-byte LDS row stride (SB=36).
__device__ __forceinline__ void tdm_load_tile(v8i g1, const unsigned short* g,
                                              unsigned int ldsOff) {
  unsigned long long ga = (unsigned long long)(size_t)g;
  v4u g0;
  g0[0] = 1u;                                        // count=1, user mode
  g0[1] = ldsOff;                                    // LDS byte address
  g0[2] = (unsigned int)ga;                          // global addr [31:0]
  g0[3] = (unsigned int)((ga >> 32) & 0x01FFFFFFull) // global addr [56:32]
        | 0x80000000u;                               // type=2 ("image")
  v4i z4 = {0, 0, 0, 0};                             // groups 2/3 unused (2-D)
  v8i z8 = {0, 0, 0, 0, 0, 0, 0, 0};
  __builtin_amdgcn_tensor_load_to_lds(g0, g1, z4, z4, z8, 0);
}

// ---------------------------------------------------------------------------
// Prep kernels
// ---------------------------------------------------------------------------
__global__ void prep_weights(const float* __restrict__ Wih,
                             const float* __restrict__ Whh,
                             unsigned short* __restrict__ Wb) {
  size_t idx = (size_t)blockIdx.x * blockDim.x + threadIdx.x;  // 4096*2048
  int j = (int)(idx >> 11);
  int k = (int)(idx & 2047);
  float v = (k < DH) ? Wih[(size_t)j * DH + k] : Whh[(size_t)j * DH + (k - DH)];
  Wb[idx] = f2bf(v);
}

__global__ void prep_bias(const float* __restrict__ bih,
                          const float* __restrict__ bhh,
                          float* __restrict__ bias) {
  int i = blockIdx.x * blockDim.x + threadIdx.x;   // 4096
  bias[i] = bih[i] + bhh[i];
}

__global__ void init_state(const float* __restrict__ h0, const float* __restrict__ c0,
                           const float* __restrict__ h02, const float* __restrict__ c02,
                           float* __restrict__ H1, float* __restrict__ C1,
                           float* __restrict__ H2, float* __restrict__ C2) {
  int idx = blockIdx.x * blockDim.x + threadIdx.x;  // 128*1024
  int d = idx & (DH - 1);
  H1[idx] = h0[d]; C1[idx] = c0[d]; H2[idx] = h02[d]; C2[idx] = c02[d];
}

__global__ void final_gather(const float* __restrict__ H1, const float* __restrict__ C1,
                             const float* __restrict__ H2, const float* __restrict__ C2,
                             float* __restrict__ out) {
  int idx = blockIdx.x * blockDim.x + threadIdx.x;  // 4 * 128*1024
  const int SZ = BATCH * DH;
  int seg = idx / SZ, off = idx % SZ;
  float v = (seg == 0) ? H1[off] : (seg == 1) ? C1[off] : (seg == 2) ? H2[off] : C2[off];
  out[idx] = v;
}

// ---------------------------------------------------------------------------
// One LSTM cell step: gates = [aLeft | Hin] @ Wb^T + bias, then pointwise.
// grid (16,4): x -> 64 hidden cols, y -> 32 batch rows. 256 threads = 8 waves.
// Wave w owns m-tile (w&1), 16-col group (w>>1), all 4 gate quadrants in regs.
// B tiles arrive via Tensor Data Mover (triple-buffered, 1 barrier/iter).
// ---------------------------------------------------------------------------
__global__ __launch_bounds__(256) void lstm_cell_step(
    const unsigned short* __restrict__ Wb,   // [4096][2048] bf16 bits
    const float* __restrict__ bias,          // [4096] b_ih+b_hh, gate order i,f,g,o
    const int*   __restrict__ tokens,        // x[t] : [128]
    const float* __restrict__ embA,          // non-null => gather A-left from embedding
    const float* __restrict__ dirA,          // direct A-left (layer 2: h1 candidate)
    const float* __restrict__ Hin, const float* __restrict__ Cin,
    const float* __restrict__ Hfb, const float* __restrict__ Cfb,  // pad fallback
    float* __restrict__ Hout, float* __restrict__ Cout,
    float* __restrict__ Hcand)               // unmasked h candidate (may be null)
{
  __shared__ unsigned short lA[2 * 32 * SA];    //  4.6 KB, double buffered
  __shared__ unsigned short lB[3 * 256 * SB];   // 55.3 KB, triple buffered (TDM dest)

  const int tid  = threadIdx.x;
  const int wave = tid >> 5;
  const int lane = tid & 31;
  const int j0   = blockIdx.x * 64;          // hidden col base
  const int m0   = blockIdx.y * 32;          // batch row base

  // A staging role
  const int arow = tid >> 3;                 // 0..31
  const int akq  = (tid & 7) << 2;           // 0,4,...,28

  // hoisted A-source bases (no per-iteration gather branch)
  const int tokA = tokens[m0 + arow];
  const float* aL = embA ? (embA + (size_t)tokA * DH)
                         : (dirA + (size_t)(m0 + arow) * DH);
  const float* aR = Hin + (size_t)(m0 + arow) * DH;

  // constant TDM descriptor group 1
  v8i g1;
  g1[0] = (int)((1u << 16)     // data_size = 2 bytes
              | (1u << 20)     // pad_enable
              | (3u << 22)     // pad_interval: 16 DWORDs
              | (1u << 25));   // pad_amount:    2 DWORDs  -> 72 B LDS row
  g1[1] = (int)(2048u << 16);  // tensor_dim0 = 2048 (k extent)
  g1[2] = (int)(4096u << 16);  // tensor_dim1 = 4096 (gate rows)
  g1[3] = (int)(32u << 16);    // tile_dim0 = 32
  g1[4] = 64;                  // tile_dim1 = 64, tile_dim2 = 0
  g1[5] = 2048;                // tensor_dim0_stride = 2048
  g1[6] = 0; g1[7] = 0;

  const unsigned int ldsB0 = (unsigned int)(size_t)&lB[0];
  const unsigned int bufBytes = 256 * SB * 2;

  v8f acc[4] = {};                           // quadrants i,f,g,o

  // preload B chunk 0 via TDM (wave 0 only; EXEC-independent scalar-path op)
  if (tid < 32) {
#pragma unroll
    for (int q = 0; q < 4; ++q)
      tdm_load_tile(g1, Wb + (size_t)(q * DH + j0) * KTOT,
                    ldsB0 + (unsigned int)(q * 64 * SB * 2));
  }

  for (int kc = 0; kc < NKC; ++kc) {
    const int k0 = kc * KC;
    const int abase = (kc & 1) * 32 * SA;
    const int bbase = (kc % 3) * 256 * SB;

    // ---- stage A tile (f32 -> bf16), prefetch next A chunk
    {
      const float* src = ((k0 < DH) ? (aL + k0) : (aR + (k0 - DH))) + akq;
      if (kc + 1 < NKC) {
        const int kn = k0 + KC;
        __builtin_prefetch(((kn < DH) ? (aL + kn) : (aR + (kn - DH))) + akq, 0, 1);
      }
      float4 f = *reinterpret_cast<const float4*>(src);
      unsigned int p0 = (unsigned int)f2bf(f.x) | ((unsigned int)f2bf(f.y) << 16);
      unsigned int p1 = (unsigned int)f2bf(f.z) | ((unsigned int)f2bf(f.w) << 16);
      unsigned int* d = reinterpret_cast<unsigned int*>(&lA[abase + arow * SA + akq]);
      d[0] = p0; d[1] = p1;
    }

    // ---- wave 0: issue TDM for chunk kc+1, then wait for chunk kc's 4 tiles
    if (tid < 32) {
      if (kc + 1 < NKC) {
        const int kn = k0 + KC;
        const unsigned int ldsNext = ldsB0 + (unsigned int)(((kc + 1) % 3)) * bufBytes;
#pragma unroll
        for (int q = 0; q < 4; ++q)
          tdm_load_tile(g1, Wb + (size_t)(q * DH + j0) * KTOT + kn,
                        ldsNext + (unsigned int)(q * 64 * SB * 2));
        __builtin_amdgcn_s_wait_tensorcnt(4);   // first 4 (chunk kc) complete
      } else {
        __builtin_amdgcn_s_wait_tensorcnt(0);   // last chunk complete
      }
    }
    __syncthreads();   // A stores visible + TDM data landed

    // ---- A fragment: 16x32 bf16 per ISA lane layout
    Frag fa;
    {
      const int r  = ((wave & 1) << 4) + (lane & 15);
      const int kh = lane >> 4;
#pragma unroll
      for (int i = 0; i < 8; ++i) {
        const int kk = ((i >> 2) << 4) + (kh << 3) + ((i & 3) << 1);
        fa.u[i] = *reinterpret_cast<const unsigned int*>(&lA[abase + r * SA + kk]);
      }
    }
    // ---- 4 WMMAs: same hidden 16x16 tile across gate quadrants i,f,g,o
#pragma unroll
    for (int q = 0; q < 4; ++q) {
      Frag fb;
      const int nl = (q << 6) + ((wave >> 1) << 4) + (lane & 15);
      const int kb = (lane >> 4) << 3;   // lanes 0-15: K 0-7,16-23; lanes 16-31: K 8-15,24-31
#pragma unroll
      for (int i = 0; i < 8; ++i)
        fb.u[i] = *reinterpret_cast<const unsigned int*>(
                      &lB[bbase + nl * SB + ((i >> 2) << 4) + kb + ((i & 3) << 1)]);
      acc[q] = __builtin_amdgcn_wmma_f32_16x16x32_bf16(
          false, fa.v, false, fb.v, (short)0, acc[q], false, false);
    }
  }

  // ---- pointwise LSTM update, fully in registers (each lane owns i,f,g,o)
  const int jcol  = j0 + ((wave >> 1) << 4) + (lane & 15);
  const int mbase = m0 + ((wave & 1) << 4) + ((lane >> 4) << 3);
  const float bi = bias[jcol];
  const float bf = bias[DH     + jcol];
  const float bg = bias[2 * DH + jcol];
  const float bo = bias[3 * DH + jcol];
#pragma unroll
  for (int r = 0; r < 8; ++r) {
    const int m = mbase + r;
    const size_t o = (size_t)m * DH + jcol;
    const float gi = sigm(acc[0][r] + bi);
    const float gf = sigm(acc[1][r] + bf);
    const float gg = tanhfast(acc[2][r] + bg);
    const float go = sigm(acc[3][r] + bo);
    const float cn = gf * Cin[o] + gi * gg;
    const float hn = go * tanhfast(cn);
    const bool pad = (tokens[m] == 0);
    Hout[o] = pad ? Hfb[o] : hn;
    Cout[o] = pad ? Cfb[o] : cn;
    if (Hcand) Hcand[o] = hn;
  }
}

// ---------------------------------------------------------------------------
extern "C" void kernel_launch(void* const* d_in, const int* in_sizes, int n_in,
                              void* d_out, int out_size, void* d_ws, size_t ws_size,
                              hipStream_t stream) {
  const int*   x    = (const int*)d_in[0];
  const float* emb  = (const float*)d_in[1];
  const float* Wih  = (const float*)d_in[2];
  const float* Whh  = (const float*)d_in[3];
  const float* bih  = (const float*)d_in[4];
  const float* bhh  = (const float*)d_in[5];
  const float* h0   = (const float*)d_in[6];
  const float* c0   = (const float*)d_in[7];
  const float* h02  = (const float*)d_in[8];
  const float* c02  = (const float*)d_in[9];

  // workspace: Wb(16MB bf16) | bias(16KB f32) | 9 state buffers (4.5MB)
  char* ws = (char*)d_ws;
  const size_t WB_BYTES = (size_t)4096 * KTOT * sizeof(unsigned short);
  unsigned short* Wb = (unsigned short*)ws;
  float* bias = (float*)(ws + WB_BYTES);
  float* st   = (float*)(ws + WB_BYTES + 65536);
  const size_t SZ = (size_t)BATCH * DH;   // 131072
  float* H1[2] = { st + 0 * SZ, st + 1 * SZ };
  float* C1[2] = { st + 2 * SZ, st + 3 * SZ };
  float* H2[2] = { st + 4 * SZ, st + 5 * SZ };
  float* C2[2] = { st + 6 * SZ, st + 7 * SZ };
  float* Hcand =  st + 8 * SZ;

  prep_weights<<<(4096 * KTOT) / 256, 256, 0, stream>>>(Wih, Whh, Wb);
  prep_bias<<<16, 256, 0, stream>>>(bih, bhh, bias);
  init_state<<<(BATCH * DH) / 256, 256, 0, stream>>>(h0, c0, h02, c02,
                                                     H1[0], C1[0], H2[0], C2[0]);

  dim3 grid(16, 4), blk(256);
  for (int t = 0; t < SEQ; ++t) {
    const int p = t & 1, q = p ^ 1;
    const int* tok = x + t * BATCH;
    // layer 1: A = [emb[x_t] | h1]; pad-fallback to own pre-state
    lstm_cell_step<<<grid, blk, 0, stream>>>(Wb, bias, tok, emb, nullptr,
        H1[p], C1[p], H1[p], C1[p], H1[q], C1[q], Hcand);
    // layer 2: A = [h1_cand | h2]; pad-fallback to layer-1 PRE-state
    lstm_cell_step<<<grid, blk, 0, stream>>>(Wb, bias, tok, nullptr, Hcand,
        H2[p], C2[p], H1[p], C1[p], H2[q], C2[q], nullptr);
  }

  // SEQ even: final states in buffer index 0. Output order: h, c, h2, c2.
  final_gather<<<(4 * BATCH * DH) / 256, 256, 0, stream>>>(
      H1[0], C1[0], H2[0], C2[0], (float*)d_out);
}
